// CrossAttention_59236188946603
// MI455X (gfx1250) — matrix operand order repaired and checked by reading, not compiled, not run on hardware
//
#include <hip/hip_runtime.h>
#include <hip/hip_bf16.h>

// ---------------- problem constants (from reference) ----------------
#define BB   2      // batch
#define SQ   2048   // query sequence
#define SKV  2048   // key/value sequence
#define NH   16     // heads
#define HD   64     // head dim
#define EMB  1024   // embed = NH*HD
#define NTOK (BB * SQ)   // 4096 flattened rows

typedef __attribute__((ext_vector_type(16))) _Float16 v16h;
typedef __attribute__((ext_vector_type(8)))  _Float16 v8h;
typedef __attribute__((ext_vector_type(8)))  float    v8f;
typedef __attribute__((ext_vector_type(4)))  int      v4i;

// ---------------- gfx1250 async global->LDS staging -------------------------
#if defined(__has_builtin)
#  if __has_builtin(__builtin_amdgcn_global_load_async_to_lds_b128)
#    define USE_ASYNC_BUILTIN 1
#  endif
#  if __has_builtin(__builtin_amdgcn_s_wait_asynccnt)
#    define USE_WAITASYNC_BUILTIN 1
#  endif
#endif

// builtin signature (from hipcc diagnostic): (global int4*, local int4*, Ii, Ii)
typedef __attribute__((address_space(1))) v4i  glob_v4i;
typedef __attribute__((address_space(3))) v4i  lds_v4i;
typedef __attribute__((address_space(3))) void lds_void;

// Copy 16 bytes global -> LDS through the async LDS-DMA path (ASYNCcnt),
// bypassing VGPRs (GLOBAL_LOAD_ASYNC_TO_LDS_B128, ISA Ch.10.7/15.18).
__device__ inline void async_copy16(const _Float16* gsrc, _Float16* ldst) {
#if defined(USE_ASYNC_BUILTIN)
    __builtin_amdgcn_global_load_async_to_lds_b128(
        (glob_v4i*)gsrc, (lds_v4i*)ldst, /*offset=*/0, /*cpol=*/0);
#else
    unsigned loff = (unsigned)(unsigned long long)(lds_void*)ldst; // 32-bit LDS offset
    asm volatile("global_load_async_to_lds_b128 %0, %1, off"
                 :: "v"(loff), "v"(gsrc) : "memory");
#endif
}

__device__ inline void wait_async0() {
#if defined(USE_WAITASYNC_BUILTIN)
    __builtin_amdgcn_s_wait_asynccnt(0);
#else
    asm volatile("s_wait_asynccnt 0x0" ::: "memory");
#endif
}

// D = A(16x32 f16) * B(32x16 f16) + C(16x16 f32), wave32, gfx1250
__device__ inline v8f wmma_f16(v16h a, v16h b, v8f c) {
    return __builtin_amdgcn_wmma_f32_16x16x32_f16(
        /*neg_a=*/false, a, /*neg_b=*/false, b,
        /*c_mod=*/(short)0, c, /*reuse_a=*/false, /*reuse_b=*/false);
}

// ---------------- fragment loaders (contiguous k order, K = 16*laneHi + i, same for A and B) ----------------
// A fragment 16x32 from row-major f16 [16][ld], contraction slice at k0.
// half-index i -> K = 16*laneHi + i; lane&15 = row M.
__device__ inline v16h load_a_f16(const _Float16* p, int ld, int k0, int lane) {
    int m  = lane & 15;
    int hi = (lane >> 4) & 1;
    const _Float16* row = p + m * ld + k0 + 16 * hi;
    v8h x = *(const v8h*)(row);        // K = k0+16*hi   .. +7
    v8h y = *(const v8h*)(row + 8);    // K = k0+16*hi+8 .. +15
    v16h a;
#pragma unroll
    for (int i = 0; i < 8; ++i) { a[i] = x[i]; a[i + 8] = y[i]; }
    return a;
}

// Same A fragment but sourced from row-major f32 (converted to f16 inline).
__device__ inline v16h load_a_f32(const float* p, int ld, int k0, int lane) {
    int m  = lane & 15;
    int hi = (lane >> 4) & 1;
    const float* row = p + (size_t)m * ld + k0 + 16 * hi;
    v8f x = *(const v8f*)(row);
    v8f y = *(const v8f*)(row + 8);
    v16h a;
#pragma unroll
    for (int i = 0; i < 8; ++i) {
        a[i]     = (_Float16)x[i];
        a[i + 8] = (_Float16)y[i];
    }
    return a;
}

// B fragment 32x16 where the *memory* is row-major [N][K] (i.e. W, so B = W^T).
// b[i] = B[k0+16*hi+i][n0+lane&15] = p[(n0+ln)*ld + k0+16*hi+i] -> contiguous.
__device__ inline v16h load_b_nk_f32(const float* p, int ld, int k0, int n0, int lane) {
    int n  = n0 + (lane & 15);
    int hi = (lane >> 4) & 1;
    const float* row = p + (size_t)n * ld + k0 + 16 * hi;
    v8f x = *(const v8f*)(row);
    v8f y = *(const v8f*)(row + 8);
    v16h b;
#pragma unroll
    for (int i = 0; i < 8; ++i) {
        b[i]     = (_Float16)x[i];
        b[i + 8] = (_Float16)y[i];
    }
    return b;
}

__device__ inline v16h load_b_nk_f16(const _Float16* p, int ld, int k0, int n0, int lane) {
    int n  = n0 + (lane & 15);
    int hi = (lane >> 4) & 1;
    const _Float16* row = p + n * ld + k0 + 16 * hi;
    v8h x = *(const v8h*)(row);
    v8h y = *(const v8h*)(row + 8);
    v16h b;
#pragma unroll
    for (int i = 0; i < 8; ++i) { b[i] = x[i]; b[i + 8] = y[i]; }
    return b;
}

// B fragment 32x16 from row-major [K][N] f16 (strided gather per lane).
__device__ inline v16h load_b_kn_f16(const _Float16* p, int ld, int n0, int lane) {
    int n  = n0 + (lane & 15);
    int hi = (lane >> 4) & 1;
    v16h b;
#pragma unroll
    for (int i = 0; i < 16; ++i) b[i] = p[(16 * hi + i) * ld + n];
    return b;
}

// ============================================================================
// Kernel 1: per-head projections.  v = x@Wv^T+bv ; k = v@Wk^T+bk (quirk: from
// projected v) ; q = (x@Wq^T+bq) * 1/sqrt(64).  Outputs packed f16 [NTOK,NH,HD].
// ============================================================================
__global__ __launch_bounds__(256) void proj_kernel(
    const float* __restrict__ q_in, const float* __restrict__ v_in,
    const float* __restrict__ Wv, const float* __restrict__ bv,
    const float* __restrict__ Wk, const float* __restrict__ bk,
    const float* __restrict__ Wq, const float* __restrict__ bq,
    _Float16* __restrict__ qg, _Float16* __restrict__ kg, _Float16* __restrict__ vg)
{
    __shared__ __align__(16) _Float16 vtile[8][16 * 64];  // per-wave projected-v tile

    int lane = threadIdx.x & 31;
    int wave = threadIdx.x >> 5;
    int ln   = lane & 15;
    int hi   = (lane >> 4) & 1;
    int r0   = blockIdx.x * 128 + wave * 16;   // flattened token row (b*SQ + s)

    for (int h = 0; h < NH; ++h) {
        // ---- V projection ----
        const float* vin = v_in + (size_t)r0 * EMB + h * HD;
        v16h a0 = load_a_f32(vin, EMB, 0, lane);
        v16h a1 = load_a_f32(vin, EMB, 32, lane);
#pragma unroll
        for (int n = 0; n < 4; ++n) {
            v8f c = {};
            c = wmma_f16(a0, load_b_nk_f32(Wv, HD, 0,  n * 16, lane), c);
            c = wmma_f16(a1, load_b_nk_f32(Wv, HD, 32, n * 16, lane), c);
            float bias = bv[n * 16 + ln];
#pragma unroll
            for (int r = 0; r < 8; ++r) {
                int row = r + 8 * hi;
                _Float16 hv = (_Float16)(c[r] + bias);
                vtile[wave][row * 64 + n * 16 + ln] = hv;   // for K projection
                vg[((size_t)(r0 + row) * NH + h) * HD + n * 16 + ln] = hv;
            }
        }
        // ---- K projection from the ALREADY-PROJECTED v (reference quirk) ----
        v16h va0 = load_a_f16(vtile[wave], 64, 0, lane);
        v16h va1 = load_a_f16(vtile[wave], 64, 32, lane);
#pragma unroll
        for (int n = 0; n < 4; ++n) {
            v8f c = {};
            c = wmma_f16(va0, load_b_nk_f32(Wk, HD, 0,  n * 16, lane), c);
            c = wmma_f16(va1, load_b_nk_f32(Wk, HD, 32, n * 16, lane), c);
            float bias = bk[n * 16 + ln];
#pragma unroll
            for (int r = 0; r < 8; ++r) {
                int row = r + 8 * hi;
                kg[((size_t)(r0 + row) * NH + h) * HD + n * 16 + ln] =
                    (_Float16)(c[r] + bias);
            }
        }
        // ---- Q projection (fold in 1/sqrt(HD) = 0.125) ----
        const float* qin = q_in + (size_t)r0 * EMB + h * HD;
        v16h qa0 = load_a_f32(qin, EMB, 0, lane);
        v16h qa1 = load_a_f32(qin, EMB, 32, lane);
#pragma unroll
        for (int n = 0; n < 4; ++n) {
            v8f c = {};
            c = wmma_f16(qa0, load_b_nk_f32(Wq, HD, 0,  n * 16, lane), c);
            c = wmma_f16(qa1, load_b_nk_f32(Wq, HD, 32, n * 16, lane), c);
            float bias = bq[n * 16 + ln];
#pragma unroll
            for (int r = 0; r < 8; ++r) {
                int row = r + 8 * hi;
                qg[((size_t)(r0 + row) * NH + h) * HD + n * 16 + ln] =
                    (_Float16)((c[r] + bias) * 0.125f);
            }
        }
    }
}

// ============================================================================
// Kernel 2: flash attention.  Block = (b, h, 64 q rows), 4 waves x 16 rows.
// K/V tiles staged via gfx1250 async global->LDS DMA (ASYNCcnt); online
// softmax with half-wave shuffles; O accumulated in f32 WMMA C fragments.
// ============================================================================
__global__ __launch_bounds__(128) void flash_kernel(
    const _Float16* __restrict__ qg, const _Float16* __restrict__ kg,
    const _Float16* __restrict__ vg, _Float16* __restrict__ ag)
{
    __shared__ __align__(16) _Float16 qtile[64 * 64];
    __shared__ __align__(16) _Float16 ktile[32 * 64];
    __shared__ __align__(16) _Float16 vtile[32 * 64];
    __shared__ __align__(16) _Float16 pst[4][16 * 32];   // per-wave P staging

    int t    = threadIdx.x;
    int lane = t & 31;
    int wave = t >> 5;
    int ln   = lane & 15;
    int hi   = (lane >> 4) & 1;
    int q0   = blockIdx.x * 64;
    int h    = blockIdx.y;
    int b    = blockIdx.z;

    // stage the 64x64 Q tile via async LDS-DMA (128 threads x 64B)
    {
        int row = t >> 1, part = t & 1;
        const _Float16* src = qg + ((size_t)(b * SQ + q0 + row) * NH + h) * HD + part * 32;
        _Float16* dst = qtile + row * 64 + part * 32;
#pragma unroll
        for (int i = 0; i < 4; ++i)
            async_copy16(src + i * 8, dst + i * 8);
    }
    wait_async0();
    __syncthreads();

    v16h qa0 = load_a_f16(qtile + wave * 16 * 64, 64, 0, lane);
    v16h qa1 = load_a_f16(qtile + wave * 16 * 64, 64, 32, lane);

    float m[8], l[8];
    v8f   O[4];
#pragma unroll
    for (int r = 0; r < 8; ++r) { m[r] = -1e30f; l[r] = 0.0f; }
#pragma unroll
    for (int n = 0; n < 4; ++n) { v8f z = {}; O[n] = z; }

    for (int kv0 = 0; kv0 < SKV; kv0 += 32) {
        __syncthreads();   // previous tile fully consumed by all waves
        {
            int row = t >> 2, part = t & 3;
            size_t base = ((size_t)(b * SKV + kv0 + row) * NH + h) * HD + part * 16;
            int    loff = row * 64 + part * 16;
            async_copy16(kg + base,     ktile + loff);
            async_copy16(kg + base + 8, ktile + loff + 8);
            async_copy16(vg + base,     vtile + loff);
            async_copy16(vg + base + 8, vtile + loff + 8);
        }
        wait_async0();     // this wave's ASYNCcnt == 0
        __syncthreads();   // all waves' tiles resident

        // S = Q * K^T  (ktile is [key][d] row-major, i.e. B^T -> use nk loader)
        v8f S0 = {}, S1 = {};
        S0 = wmma_f16(qa0, load_b_nk_f16(ktile, 64, 0,  0,  lane), S0);
        S0 = wmma_f16(qa1, load_b_nk_f16(ktile, 64, 32, 0,  lane), S0);
        S1 = wmma_f16(qa0, load_b_nk_f16(ktile, 64, 0,  16, lane), S1);
        S1 = wmma_f16(qa1, load_b_nk_f16(ktile, 64, 32, 16, lane), S1);

        // online softmax; each C row lives in one 16-lane half -> xor 1,2,4,8
        float alpha[8];
#pragma unroll
        for (int r = 0; r < 8; ++r) {
            float tm = fmaxf(S0[r], S1[r]);
            tm = fmaxf(tm, __shfl_xor(tm, 1, 32));
            tm = fmaxf(tm, __shfl_xor(tm, 2, 32));
            tm = fmaxf(tm, __shfl_xor(tm, 4, 32));
            tm = fmaxf(tm, __shfl_xor(tm, 8, 32));
            float mn = fmaxf(m[r], tm);
            alpha[r] = __expf(m[r] - mn);
            m[r] = mn;
            float p0 = __expf(S0[r] - mn);
            float p1 = __expf(S1[r] - mn);
            float rs = p0 + p1;
            rs += __shfl_xor(rs, 1, 32);
            rs += __shfl_xor(rs, 2, 32);
            rs += __shfl_xor(rs, 4, 32);
            rs += __shfl_xor(rs, 8, 32);
            l[r] = l[r] * alpha[r] + rs;
            int row = r + 8 * hi;
            pst[wave][row * 32 + ln]      = (_Float16)p0;
            pst[wave][row * 32 + 16 + ln] = (_Float16)p1;
        }
#pragma unroll
        for (int n = 0; n < 4; ++n)
#pragma unroll
            for (int r = 0; r < 8; ++r) O[n][r] *= alpha[r];

        // O += P * V   (P via LDS round-trip to get A layout)
        v16h pa = load_a_f16(pst[wave], 32, 0, lane);
#pragma unroll
        for (int n = 0; n < 4; ++n)
            O[n] = wmma_f16(pa, load_b_kn_f16(vtile, 64, n * 16, lane), O[n]);
    }

    // normalize and write attention output f16 [B*SQ, EMB]
#pragma unroll
    for (int n = 0; n < 4; ++n) {
#pragma unroll
        for (int r = 0; r < 8; ++r) {
            int row = q0 + wave * 16 + r + 8 * hi;
            float val = O[n][r] / l[r];
            ag[((size_t)(b * SQ) + row) * EMB + h * HD + n * 16 + ln] = (_Float16)val;
        }
    }
}

// ============================================================================
// Kernel 3: output dense  Y[4096,1024] = A @ Wd^T + bd  (f32 out)
// ============================================================================
__global__ __launch_bounds__(256) void dense_kernel(
    const _Float16* __restrict__ ag, const float* __restrict__ Wd,
    const float* __restrict__ bd, float* __restrict__ out)
{
    int lane = threadIdx.x & 31;
    int wave = threadIdx.x >> 5;
    int ln   = lane & 15;
    int hi   = (lane >> 4) & 1;
    int r0   = blockIdx.x * 128 + wave * 16;
    int c0   = blockIdx.y * 64;

    v8f acc[4];
#pragma unroll
    for (int n = 0; n < 4; ++n) { v8f z = {}; acc[n] = z; }

    for (int k0 = 0; k0 < EMB; k0 += 32) {
        v16h a = load_a_f16(ag + (size_t)r0 * EMB, EMB, k0, lane);
#pragma unroll
        for (int n = 0; n < 4; ++n)
            acc[n] = wmma_f16(a, load_b_nk_f32(Wd, EMB, k0, c0 + n * 16, lane), acc[n]);
    }
#pragma unroll
    for (int n = 0; n < 4; ++n) {
        float bias = bd[c0 + n * 16 + ln];
#pragma unroll
        for (int r = 0; r < 8; ++r) {
            int row = r0 + r + 8 * hi;
            out[(size_t)row * EMB + c0 + n * 16 + ln] = acc[n][r] + bias;
        }
    }
}

// ============================================================================
extern "C" void kernel_launch(void* const* d_in, const int* in_sizes, int n_in,
                              void* d_out, int out_size, void* d_ws, size_t ws_size,
                              hipStream_t stream) {
    const float* queries = (const float*)d_in[0];
    const float* values  = (const float*)d_in[1];
    // d_in[2] = heads (constant 16, unused)
    const float* Wv = (const float*)d_in[3];
    const float* bv = (const float*)d_in[4];
    const float* Wk = (const float*)d_in[5];
    const float* bk = (const float*)d_in[6];
    const float* Wq = (const float*)d_in[7];
    const float* bq = (const float*)d_in[8];
    const float* Wd = (const float*)d_in[9];
    const float* bd = (const float*)d_in[10];
    float* out = (float*)d_out;

    const size_t PLANE = (size_t)NTOK * EMB;     // 4,194,304 halves (8 MB)
    _Float16* qg = (_Float16*)d_ws;              // [NTOK, NH, HD] f16
    _Float16* kg = qg + PLANE;
    _Float16* vg = kg + PLANE;
    _Float16* ag = vg + PLANE;                   // attention output [NTOK, EMB]

    proj_kernel<<<dim3(NTOK / 128), 256, 0, stream>>>(
        queries, values, Wv, bv, Wk, bk, Wq, bq, qg, kg, vg);
    flash_kernel<<<dim3(SQ / 64, NH, BB), 128, 0, stream>>>(qg, kg, vg, ag);
    dense_kernel<<<dim3(NTOK / 128, EMB / 64), 256, 0, stream>>>(ag, Wd, bd, out);
}
